// AttentionBlock_17489106830009
// MI455X (gfx1250) — matrix operand order repaired and checked by reading, not compiled, not run on hardware
//
#include <hip/hip_runtime.h>
#include <hip/hip_bf16.h>

typedef __attribute__((ext_vector_type(16))) _Float16 v16h;
typedef __attribute__((ext_vector_type(8)))  float    v8f;

#define B_   4
#define C_   512
#define T_   2048
#define HEADS_ 8
#define CH_  64          // C/HEADS
#define BH_  32          // B*HEADS
#define GROUPS_ 32
#define GN_ELEMS (16 * T_)   // (C/GROUPS) * T
#define LDSP 40              // padded LDS pitch (halves) for 32-wide K tiles

union FragH { v16h v; unsigned int u[8]; };

// ---- CDNA5 async-to-LDS helpers (ASYNCcnt path) ----
__device__ __forceinline__ void async_b128(unsigned lds_off, const void* gptr) {
    asm volatile("global_load_async_to_lds_b128 %0, %1, off"
                 :: "v"(lds_off), "v"(gptr) : "memory");
}
__device__ __forceinline__ void wait_async0() {
    asm volatile("s_wait_asynccnt 0" ::: "memory");
}
__device__ __forceinline__ unsigned lds_off(const void* p) {
    return (unsigned)(uintptr_t)p;   // low 32 bits of flat shared addr = LDS offset
}

// A fragment 16x32 f16 (MxK). Storage: base[m][k], k contiguous, pitch in halves.
// lane<16: M=lane, K in {0..7}U{16..23}; lane>=16: M=lane-16, K in {8..15}U{24..31}
__device__ __forceinline__ v16h load_frag_a(const _Float16* base, int pitch, int lane) {
    FragH f;
    const _Float16* row = base + (lane & 15) * pitch;
    int kb = (lane >> 4) * 8;
#pragma unroll
    for (int j = 0; j < 4; ++j) f.u[j]     = *(const unsigned int*)(row + kb + 2 * j);
#pragma unroll
    for (int j = 0; j < 4; ++j) f.u[4 + j] = *(const unsigned int*)(row + 16 + kb + 2 * j);
    return f.v;
}

// B fragment 32x16 f16 (KxN) from TRANSPOSED storage baseT[n][k], k contiguous.
// lane<16: N=lane, K in {0..7}U{16..23}; lane>=16: N=lane-16, K in {8..15}U{24..31}
__device__ __forceinline__ v16h load_frag_bT(const _Float16* baseT, int pitch, int lane) {
    FragH f;
    const _Float16* row = baseT + (lane & 15) * pitch;
    int kb = (lane >> 4) * 8;
#pragma unroll
    for (int j = 0; j < 4; ++j) f.u[j]     = *(const unsigned int*)(row + kb + 2 * j);
#pragma unroll
    for (int j = 0; j < 4; ++j) f.u[4 + j] = *(const unsigned int*)(row + 16 + kb + 2 * j);
    return f.v;
}

// ---------------- GroupNorm (fp32 stats, f16 output) ----------------
__global__ __launch_bounds__(256) void groupnorm_kernel(const float* __restrict__ x,
                                                        const float* __restrict__ gw,
                                                        const float* __restrict__ gb,
                                                        _Float16* __restrict__ h) {
    int grp = blockIdx.x;                       // b*32 + g
    int tid = threadIdx.x;
    const float* xg = x + (size_t)grp * GN_ELEMS;
    _Float16* hg = h + (size_t)grp * GN_ELEMS;
    int cbase = (grp & 31) * 16;
    __shared__ float s_sum[256], s_sq[256];
    float sum = 0.f, sq = 0.f;
    for (int i = tid; i < GN_ELEMS / 4; i += 256) {
        float4 v = ((const float4*)xg)[i];
        sum += v.x + v.y + v.z + v.w;
        sq  += v.x * v.x + v.y * v.y + v.z * v.z + v.w * v.w;
    }
    s_sum[tid] = sum; s_sq[tid] = sq;
    __syncthreads();
    for (int s = 128; s > 0; s >>= 1) {
        if (tid < s) { s_sum[tid] += s_sum[tid + s]; s_sq[tid] += s_sq[tid + s]; }
        __syncthreads();
    }
    float mu   = s_sum[0] * (1.0f / GN_ELEMS);
    float var  = s_sq[0]  * (1.0f / GN_ELEMS) - mu * mu;
    float rstd = rsqrtf(var + 1e-5f);
    for (int i = tid; i < GN_ELEMS; i += 256) {
        int c = cbase + (i >> 11);
        float v = (xg[i] - mu) * rstd * gw[c] + gb[c];
        hg[i] = (_Float16)v;
    }
}

// ---------------- weight f32 -> f16 convert ----------------
__global__ __launch_bounds__(256) void cvt_kernel(const float* __restrict__ qkvw,
                                                  const float* __restrict__ projw,
                                                  _Float16* __restrict__ wq,
                                                  _Float16* __restrict__ wp) {
    int i = blockIdx.x * 256 + threadIdx.x;
    if (i < 3 * C_ * C_) wq[i] = (_Float16)qkvw[i];
    if (i < C_ * C_)     wp[i] = (_Float16)projw[i];
}

// ---------------- shared WMMA GEMM mainloop: 128x128 block, 8 waves ----------------
// Double-buffered: A tile via async-to-LDS, B tile register-staged + LDS transpose.
// Wave grid 4(M)x2(N): each wave owns 32(M)x64(N) = 2x4 accumulator tiles.
__device__ __forceinline__ void gemm_tile(const _Float16* Ag, int aPitch,   // MxK row-major
                                          const _Float16* Bg, int bPitch,   // KxN row-major
                                          int ksteps,
                                          _Float16 (*aL)[128][LDSP],        // [2][128][LDSP]
                                          _Float16 (*bL)[128][LDSP],        // transposed [n][k]
                                          v8f acc[2][4]) {
    const int tid  = threadIdx.x;
    const int lane = tid & 31;
    const int wid  = tid >> 5;
    const int wm   = (wid >> 1) * 32;
    const int wn   = (wid & 1) * 64;
    const int arow = tid >> 1, acol = (tid & 1) * 16;   // A: 32B per thread
    uint4 breg[2];

    auto issueA = [&](int ks, int buf) {
        const _Float16* src = Ag + (size_t)arow * aPitch + ks * 32 + acol;
        async_b128(lds_off(&aL[buf][arow][acol]), src);
        async_b128(lds_off(&aL[buf][arow][acol + 8]), src + 8);
    };
    auto loadB = [&](int ks) {
        const _Float16* bk = Bg + (size_t)(ks * 32) * bPitch;
#pragma unroll
        for (int c = 0; c < 2; ++c) {
            int chunk = tid * 2 + c;
            int k = chunk >> 4, n = (chunk & 15) * 8;
            breg[c] = *(const uint4*)(bk + (size_t)k * bPitch + n);
        }
    };
    auto storeB = [&](int buf) {
#pragma unroll
        for (int c = 0; c < 2; ++c) {
            int chunk = tid * 2 + c;
            int k = chunk >> 4, n = (chunk & 15) * 8;
            const _Float16* hp = (const _Float16*)&breg[c];
#pragma unroll
            for (int i = 0; i < 8; ++i) bL[buf][n + i][k] = hp[i];
        }
    };

    issueA(0, 0);
    loadB(0);
    storeB(0);
    wait_async0();
    __syncthreads();                       // tile 0 resident
    for (int ks = 0; ks < ksteps; ++ks) {
        int buf = ks & 1;
        if (ks + 1 < ksteps) {             // prefetch next tile while computing
            issueA(ks + 1, buf ^ 1);
            loadB(ks + 1);
        }
        v16h afrag[2], bfrag[4];
#pragma unroll
        for (int mt = 0; mt < 2; ++mt) afrag[mt] = load_frag_a(&aL[buf][wm + mt * 16][0], LDSP, lane);
#pragma unroll
        for (int nt = 0; nt < 4; ++nt) bfrag[nt] = load_frag_bT(&bL[buf][wn + nt * 16][0], LDSP, lane);
#pragma unroll
        for (int mt = 0; mt < 2; ++mt)
#pragma unroll
            for (int nt = 0; nt < 4; ++nt)
                acc[mt][nt] = __builtin_amdgcn_wmma_f32_16x16x32_f16(
                    false, afrag[mt], false, bfrag[nt], (short)0, acc[mt][nt], false, false);
        __syncthreads();                   // all waves done reading buf
        if (ks + 1 < ksteps) {
            storeB(buf ^ 1);
            wait_async0();                 // A(ks+1) landed in LDS
            __syncthreads();               // tile ks+1 published
        }
    }
}

// ---------------- QKV GEMM: (3C x C) @ (C x T), scatter to q/k (transposed, scaled) and v --------
__global__ __launch_bounds__(256) void qkv_kernel(const _Float16* __restrict__ wq,
                                                  const _Float16* __restrict__ h,
                                                  const float* __restrict__ qkv_b,
                                                  _Float16* __restrict__ qT,
                                                  _Float16* __restrict__ kT,
                                                  _Float16* __restrict__ vbuf) {
    __shared__ _Float16 aL[2][128][LDSP];
    __shared__ _Float16 bL[2][128][LDSP];
    int bm = blockIdx.x * 128, bn = blockIdx.y * 128, b = blockIdx.z;
    v8f acc[2][4] = {};
    gemm_tile(wq + (size_t)bm * C_, C_, h + ((size_t)b * C_) * T_ + bn, T_, C_ / 32, aL, bL, acc);
    const int lane = threadIdx.x & 31, wid = threadIdx.x >> 5;
    const int wm = (wid >> 1) * 32, wn = (wid & 1) * 64;
    const float SCALE = 0.35355339059327373f;  // 1/sqrt(sqrt(64))
#pragma unroll
    for (int mt = 0; mt < 2; ++mt)
#pragma unroll
        for (int nt = 0; nt < 4; ++nt) {
            int obase = bm + wm + mt * 16 + ((lane >> 4) << 3);  // 8-aligned, inside one 64-block
            int t = bn + wn + nt * 16 + (lane & 15);
            int hd = obase / 192, rm0 = obase % 192;             // uniform over the 8 rows
            size_t bh = (size_t)b * HEADS_ + hd;
            if (rm0 < 128) {                                     // q or k: contiguous ch -> b128 store
                union { uint4 q; _Float16 hh[8]; } pack;
#pragma unroll
                for (int r = 0; r < 8; ++r)
                    pack.hh[r] = (_Float16)((acc[mt][nt][r] + qkv_b[obase + r]) * SCALE);
                if (rm0 < 64)
                    *(uint4*)(qT + (bh * T_ + t) * CH_ + rm0) = pack.q;
                else
                    *(uint4*)(kT + (bh * T_ + t) * CH_ + (rm0 - 64)) = pack.q;
            } else {                                             // v: [ch][t] scatter
#pragma unroll
                for (int r = 0; r < 8; ++r)
                    vbuf[(bh * CH_ + (rm0 - 128 + r)) * T_ + t] =
                        (_Float16)(acc[mt][nt][r] + qkv_b[obase + r]);
            }
        }
}

// ---------------- Flash attention: one block = (bh, 128 query rows); wave = 16 rows ----------------
__global__ __launch_bounds__(256) void attn_kernel(const _Float16* __restrict__ qT,
                                                   const _Float16* __restrict__ kT,
                                                   const _Float16* __restrict__ vbuf,
                                                   _Float16* __restrict__ aout) {
    __shared__ _Float16 kL[2][32][72];      // [s][ch]
    __shared__ _Float16 vL[2][64][40];      // [ch][s]
    __shared__ _Float16 pL[8][16][40];      // per-wave [t][s]
    __shared__ float bcast[8][16];
    int bh = blockIdx.x, tb = blockIdx.y;
    int tid = threadIdx.x, lane = tid & 31, wid = tid >> 5;
    int t0 = tb * 128 + wid * 16;
    const _Float16* qBase = qT + ((size_t)bh * T_ + t0) * CH_;
    v16h qfrag[2];
    qfrag[0] = load_frag_a(qBase, CH_, lane);        // ch 0..31
    qfrag[1] = load_frag_a(qBase + 32, CH_, lane);   // ch 32..63
    v8f oacc[4] = {};
    float m_r[8], l_r[8];
#pragma unroll
    for (int r = 0; r < 8; ++r) { m_r[r] = -1e30f; l_r[r] = 0.f; }

    const int krow = tid >> 3, kch = (tid & 7) * 8;  // 16B per thread, each tile
    const int vch  = tid >> 2, vsc = (tid & 3) * 8;
    auto issueKV = [&](int s0, int buf) {
        async_b128(lds_off(&kL[buf][krow][kch]),
                   kT + ((size_t)bh * T_ + s0 + krow) * CH_ + kch);
        async_b128(lds_off(&vL[buf][vch][vsc]),
                   vbuf + ((size_t)bh * CH_ + vch) * T_ + s0 + vsc);
    };
    issueKV(0, 0);
    wait_async0();
    __syncthreads();

    for (int s0 = 0; s0 < T_; s0 += 32) {
        int buf = (s0 >> 5) & 1;
        if (s0 + 32 < T_) issueKV(s0 + 32, buf ^ 1);   // prefetch next K/V tile
        // S = Q^T K : two 16x16 s-subtiles, K-depth 64 = 2 WMMAs each
        v8f sAcc[2];
#pragma unroll
        for (int ss = 0; ss < 2; ++ss) {
            v8f c = {};
            c = __builtin_amdgcn_wmma_f32_16x16x32_f16(false, qfrag[0], false,
                    load_frag_bT(&kL[buf][ss * 16][0], 72, lane), (short)0, c, false, false);
            c = __builtin_amdgcn_wmma_f32_16x16x32_f16(false, qfrag[1], false,
                    load_frag_bT(&kL[buf][ss * 16][32], 72, lane), (short)0, c, false, false);
            sAcc[ss] = c;
        }
        // online softmax: row t = r + 8*(lane>>4); 16-lane reductions over s
        float alpha[8];
#pragma unroll
        for (int r = 0; r < 8; ++r) {
            float mx = fmaxf(sAcc[0][r], sAcc[1][r]);
#pragma unroll
            for (int m = 8; m >= 1; m >>= 1) mx = fmaxf(mx, __shfl_xor(mx, m, 32));
            float mnew = fmaxf(m_r[r], mx);
            alpha[r] = __expf(m_r[r] - mnew);
            float p0 = __expf(sAcc[0][r] - mnew);
            float p1 = __expf(sAcc[1][r] - mnew);
            sAcc[0][r] = p0; sAcc[1][r] = p1;
            float rs = p0 + p1;
#pragma unroll
            for (int m = 8; m >= 1; m >>= 1) rs += __shfl_xor(rs, m, 32);
            l_r[r] = l_r[r] * alpha[r] + rs;
            m_r[r] = mnew;
        }
        if ((lane & 15) == 0) {
#pragma unroll
            for (int r = 0; r < 8; ++r) bcast[wid][r + ((lane >> 4) << 3)] = alpha[r];
        }
#pragma unroll
        for (int ss = 0; ss < 2; ++ss)
#pragma unroll
            for (int r = 0; r < 8; ++r)
                pL[wid][r + ((lane >> 4) << 3)][ss * 16 + (lane & 15)] = (_Float16)sAcc[ss][r];
        asm volatile("s_wait_dscnt 0" ::: "memory");  // same-wave LDS transpose RAW
        float af = bcast[wid][lane & 15];
#pragma unroll
        for (int mt = 0; mt < 4; ++mt)
#pragma unroll
            for (int r = 0; r < 8; ++r) oacc[mt][r] *= af;
        // O += V @ P^T : 4 M-tiles, K-depth 32
        v16h pfrag = load_frag_bT(&pL[wid][0][0], 40, lane);
#pragma unroll
        for (int mt = 0; mt < 4; ++mt) {
            v16h vfrag = load_frag_a(&vL[buf][mt * 16][0], 40, lane);
            oacc[mt] = __builtin_amdgcn_wmma_f32_16x16x32_f16(
                false, vfrag, false, pfrag, (short)0, oacc[mt], false, false);
        }
        __syncthreads();                    // all waves done reading buf
        if (s0 + 32 < T_) {
            wait_async0();                  // next tile landed
            __syncthreads();                // publish
        }
    }
    if ((lane & 15) == 0) {
#pragma unroll
        for (int r = 0; r < 8; ++r) bcast[wid][r + ((lane >> 4) << 3)] = l_r[r];
    }
    asm volatile("s_wait_dscnt 0" ::: "memory");
    float inv = 1.0f / bcast[wid][lane & 15];
    int b = bh >> 3, hd = bh & 7;
    int t = t0 + (lane & 15);
#pragma unroll
    for (int mt = 0; mt < 4; ++mt) {
        int chbase = mt * 16 + ((lane >> 4) << 3);
#pragma unroll
        for (int r = 0; r < 8; ++r) {
            int c = hd * CH_ + chbase + r;
            aout[((size_t)b * C_ + c) * T_ + t] = (_Float16)(oacc[mt][r] * inv);
        }
    }
}

// ---------------- proj GEMM + bias + residual, fp32 out ----------------
__global__ __launch_bounds__(256) void proj_kernel(const _Float16* __restrict__ wp,
                                                   const _Float16* __restrict__ a,
                                                   const float* __restrict__ proj_b,
                                                   const float* __restrict__ x,
                                                   float* __restrict__ out) {
    __shared__ _Float16 aL[2][128][LDSP];
    __shared__ _Float16 bL[2][128][LDSP];
    int bm = blockIdx.x * 128, bn = blockIdx.y * 128, b = blockIdx.z;
    v8f acc[2][4] = {};
    gemm_tile(wp + (size_t)bm * C_, C_, a + ((size_t)b * C_) * T_ + bn, T_, C_ / 32, aL, bL, acc);
    const int lane = threadIdx.x & 31, wid = threadIdx.x >> 5;
    const int wm = (wid >> 1) * 32, wn = (wid & 1) * 64;
#pragma unroll
    for (int mt = 0; mt < 2; ++mt)
#pragma unroll
        for (int nt = 0; nt < 4; ++nt) {
            int obase = bm + wm + mt * 16 + ((lane >> 4) << 3);
            int t = bn + wn + nt * 16 + (lane & 15);
#pragma unroll
            for (int r = 0; r < 8; ++r) {
                int o = obase + r;
                size_t idx = ((size_t)b * C_ + o) * T_ + t;
                out[idx] = x[idx] + acc[mt][nt][r] + proj_b[o];
            }
        }
}

extern "C" void kernel_launch(void* const* d_in, const int* in_sizes, int n_in,
                              void* d_out, int out_size, void* d_ws, size_t ws_size,
                              hipStream_t stream) {
    const float* x      = (const float*)d_in[0];
    const float* norm_w = (const float*)d_in[1];
    const float* norm_b = (const float*)d_in[2];
    const float* qkv_w  = (const float*)d_in[3];
    const float* qkv_b  = (const float*)d_in[4];
    const float* proj_w = (const float*)d_in[5];
    const float* proj_b = (const float*)d_in[6];
    float* out = (float*)d_out;

    char* ws = (char*)d_ws;
    const size_t SZ_BCT = (size_t)B_ * C_ * T_ * sizeof(_Float16);   // 8 MB
    _Float16* h    = (_Float16*)(ws);
    _Float16* wq   = (_Float16*)(ws + SZ_BCT);
    _Float16* wp   = (_Float16*)(ws + SZ_BCT + 1572864);
    _Float16* qT   = (_Float16*)(ws + SZ_BCT + 2097152);
    _Float16* kT   = (_Float16*)(ws + 2 * SZ_BCT + 2097152);
    _Float16* vbuf = (_Float16*)(ws + 3 * SZ_BCT + 2097152);
    _Float16* aws  = (_Float16*)(ws + 4 * SZ_BCT + 2097152);

    groupnorm_kernel<<<B_ * GROUPS_, 256, 0, stream>>>(x, norm_w, norm_b, h);
    cvt_kernel<<<(3 * C_ * C_ + 255) / 256, 256, 0, stream>>>(qkv_w, proj_w, wq, wp);
    qkv_kernel<<<dim3(12, 16, B_), 256, 0, stream>>>(wq, h, qkv_b, qT, kT, vbuf);
    attn_kernel<<<dim3(BH_, 16), 256, 0, stream>>>(qT, kT, vbuf, aws);
    proj_kernel<<<dim3(4, 16, B_), 256, 0, stream>>>(wp, aws, proj_b, x, out);
}
